// GraphNCA_67061619360163
// MI455X (gfx1250) — compile-verified, matrix operand order checked by reading, not run on hardware
//
#include <hip/hip_runtime.h>

typedef __attribute__((ext_vector_type(16))) _Float16 v16h;
typedef __attribute__((ext_vector_type(8)))  float    v8f;

#define N_NODES 100000
#define N_EDGES 3200000
#define C_CH    16
#define PC_CH   48
#define HID_CH  32
#define NTILES  (N_NODES / 16)   // 6250 16-node tiles

union HV { v16h v; _Float16 h[16]; };

// ---------------------------------------------------------------- init
__global__ void init_kernel(float* __restrict__ deg, float* __restrict__ agg) {
    int t = blockIdx.x * 256 + threadIdx.x;
    if (t < N_NODES * PC_CH) agg[t] = 0.0f;
    if (t < N_NODES)         deg[t] = 1.0f;   // self-loop contribution
}

// ---------------------------------------------------------------- degree
__global__ void degree_kernel(const int* __restrict__ ei, float* __restrict__ deg) {
    int t = blockIdx.x * 256 + threadIdx.x;
    if (t >= N_EDGES) return;
    int col = ei[N_EDGES + t];            // edge_index[1][t] (dst)
    atomicAdd(&deg[col], 1.0f);
}

// ---------------------------------------------------------------- dis = rsqrt(deg)
__global__ void rsqrt_kernel(float* __restrict__ deg) {
    int t = blockIdx.x * 256 + threadIdx.x;
    if (t < N_NODES) deg[t] = rsqrtf(deg[t]);
}

// ---------------------------------------------------------------- h = x @ W_gcn^T  (WMMA)
// A tile: 16 nodes x 16 ch (K padded 16->32).  B: W_gcn^T, 3 tiles of 16 cols.
__global__ void transform_kernel(const float* __restrict__ x,
                                 const float* __restrict__ Wg,   // [48,16]
                                 float* __restrict__ h)          // [N,48]
{
    int wave = blockIdx.x * 8 + (threadIdx.x >> 5);
    if (wave >= NTILES) return;                 // wave-uniform: EXEC stays all-ones
    int lane  = threadIdx.x & 31;
    int row   = lane & 15;                      // A: M = lane&15
    int koff  = (lane >> 4) * 8;                // lanes 16-31 cover K+8
    int col   = lane & 15;                      // B/C: N = lane&15
    int rbase = (lane >> 4) * 8;                // C: lanes 16-31 hold M=8..15

    HV a;
    const float* xp = x + (size_t)(wave * 16 + row) * C_CH + koff;
#pragma unroll
    for (int m = 0; m < 8; ++m) a.h[m] = (_Float16)xp[m];
#pragma unroll
    for (int m = 8; m < 16; ++m) a.h[m] = (_Float16)0.0f;   // K=16..31 pad

#pragma unroll
    for (int t = 0; t < 3; ++t) {               // 3 output tiles of 16 cols
        HV b;
        const float* wp = Wg + (size_t)(t * 16 + col) * C_CH + koff;  // B[k][p] = Wg[p][k]
#pragma unroll
        for (int m = 0; m < 8; ++m)  b.h[m]     = (_Float16)wp[m];
#pragma unroll
        for (int m = 0; m < 8; ++m)  b.h[8 + m] = (_Float16)0.0f;
        v8f c = {};
        c = __builtin_amdgcn_wmma_f32_16x16x32_f16(false, a.v, false, b.v,
                                                   (short)0, c, false, false);
        float* hp = h + (size_t)(wave * 16 + rbase) * PC_CH + t * 16 + col;
#pragma unroll
        for (int r = 0; r < 8; ++r) hp[(size_t)r * PC_CH] = c[r];
    }
}

// ---------------------------------------------------------------- edge scatter (L2 atomics)
__global__ void scatter_kernel(const int* __restrict__ ei,
                               const float* __restrict__ h,
                               const float* __restrict__ dis,
                               float* __restrict__ agg)
{
    long long t = (long long)blockIdx.x * 256 + threadIdx.x;
    if (t >= (long long)N_EDGES * PC_CH) return;
    int e = (int)(t / PC_CH);
    int p = (int)(t - (long long)e * PC_CH);
    int row = ei[e];
    int col = ei[N_EDGES + e];
    float norm = dis[row] * dis[col];
    atomicAdd(&agg[(size_t)col * PC_CH + p], h[(size_t)row * PC_CH + p] * norm);
}

// ---------------------------------------------------------------- fused: self-loop + MLP + residual
__global__ void fused_update_kernel(const float* __restrict__ xx,
                                    const float* __restrict__ hbuf,
                                    const float* __restrict__ agg,
                                    const float* __restrict__ dis,
                                    const float* __restrict__ W1,  // [32,48]
                                    const float* __restrict__ b1,  // [32]
                                    const float* __restrict__ W2,  // [16,32]
                                    const float* __restrict__ b2,  // [16]
                                    float* __restrict__ out)       // [N,16]
{
    __shared__ _Float16 lds[8 * 16 * HID_CH];   // per-wave 16x32 f16 staging
    int wslot = threadIdx.x >> 5;
    int wave  = blockIdx.x * 8 + wslot;
    if (wave >= NTILES) return;                 // wave-uniform
    int lane  = threadIdx.x & 31;
    int row   = lane & 15;
    int koff  = (lane >> 4) * 8;
    int col   = lane & 15;
    int rbase = (lane >> 4) * 8;

    int node = wave * 16 + row;
    float d  = dis[node];
    float d2 = d * d;
    const float* ap = agg  + (size_t)node * PC_CH;
    const float* hp = hbuf + (size_t)node * PC_CH;

    // feat = agg + h*dis^2, packed straight into A layout (K=48 padded to 64)
    HV a0, a1;
#pragma unroll
    for (int m = 0; m < 8; ++m) {
        a0.h[m]     = (_Float16)(ap[koff + m]      + hp[koff + m]      * d2);  // K 0..15
        a0.h[8 + m] = (_Float16)(ap[16 + koff + m] + hp[16 + koff + m] * d2);  // K 16..31
        a1.h[m]     = (_Float16)(ap[32 + koff + m] + hp[32 + koff + m] * d2);  // K 32..47
        a1.h[8 + m] = (_Float16)0.0f;                                          // K 48..63 pad
    }

    _Float16* myl = lds + wslot * 16 * HID_CH;

#pragma unroll
    for (int jt = 0; jt < 2; ++jt) {            // two 16-col tiles of the 32 hidden units
        const float* w1p = W1 + (size_t)(jt * 16 + col) * PC_CH;  // B[p][j] = W1[j][p]
        HV b;
#pragma unroll
        for (int m = 0; m < 8; ++m) {
            b.h[m]     = (_Float16)w1p[koff + m];        // K 0..15
            b.h[8 + m] = (_Float16)w1p[16 + koff + m];   // K 16..31
        }
        v8f c = {};
        c = __builtin_amdgcn_wmma_f32_16x16x32_f16(false, a0.v, false, b.v,
                                                   (short)0, c, false, false);
#pragma unroll
        for (int m = 0; m < 8; ++m) {
            b.h[m]     = (_Float16)w1p[32 + koff + m];   // K 32..47
            b.h[8 + m] = (_Float16)0.0f;                 // K 48..63 pad
        }
        c = __builtin_amdgcn_wmma_f32_16x16x32_f16(false, a1.v, false, b.v,
                                                   (short)0, c, false, false);

        float bias = b1[jt * 16 + col];
#pragma unroll
        for (int r = 0; r < 8; ++r) {
            float v = c[r] + bias;
            v = v > 0.0f ? v : 0.0f;             // ReLU on f32 accumulator
            myl[(rbase + r) * HID_CH + jt * 16 + col] = (_Float16)v;
        }
    }

    // Re-stage hidden (C layout -> A layout) via LDS; same-wave DS ops are in order.
    HV ah;
#pragma unroll
    for (int m = 0; m < 8; ++m) {
        ah.h[m]     = myl[row * HID_CH + koff + m];        // K 0..15
        ah.h[8 + m] = myl[row * HID_CH + 16 + koff + m];   // K 16..31
    }
    HV bw;
#pragma unroll
    for (int m = 0; m < 8; ++m) {
        bw.h[m]     = (_Float16)W2[(size_t)col * HID_CH + koff + m];       // B[j][c]=W2[c][j]
        bw.h[8 + m] = (_Float16)W2[(size_t)col * HID_CH + 16 + koff + m];
    }
    v8f co = {};
    co = __builtin_amdgcn_wmma_f32_16x16x32_f16(false, ah.v, false, bw.v,
                                                (short)0, co, false, false);
    float bo = b2[col];
#pragma unroll
    for (int r = 0; r < 8; ++r) {
        size_t idx = (size_t)(wave * 16 + rbase + r) * C_CH + col;
        out[idx] = xx[idx] + co[r] + bo;
    }
}

// ---------------------------------------------------------------- launcher
extern "C" void kernel_launch(void* const* d_in, const int* in_sizes, int n_in,
                              void* d_out, int out_size, void* d_ws, size_t ws_size,
                              hipStream_t stream) {
    (void)in_sizes; (void)n_in; (void)out_size; (void)ws_size;
    const float* xx = (const float*)d_in[0];
    const int*   ei = (const int*)d_in[1];    // edge_index [2,E]
    // d_in[2] = parent_index (unused scalar)
    const float* Wg = (const float*)d_in[3];
    const float* W1 = (const float*)d_in[4];
    const float* b1 = (const float*)d_in[5];
    const float* W2 = (const float*)d_in[6];
    const float* b2 = (const float*)d_in[7];
    float* out = (float*)d_out;

    float* deg = (float*)d_ws;                          // N floats (becomes dis)
    float* h   = deg + N_NODES;                         // N*48 floats
    float* agg = h + (size_t)N_NODES * PC_CH;           // N*48 floats

    init_kernel<<<(N_NODES * PC_CH + 255) / 256, 256, 0, stream>>>(deg, agg);
    degree_kernel<<<(N_EDGES + 255) / 256, 256, 0, stream>>>(ei, deg);
    rsqrt_kernel<<<(N_NODES + 255) / 256, 256, 0, stream>>>(deg);
    transform_kernel<<<(NTILES + 7) / 8, 256, 0, stream>>>(xx, Wg, h);
    long long sc = (long long)N_EDGES * PC_CH;
    scatter_kernel<<<(unsigned)((sc + 255) / 256), 256, 0, stream>>>(ei, h, deg, agg);
    fused_update_kernel<<<(NTILES + 7) / 8, 256, 0, stream>>>(xx, h, agg, deg,
                                                              W1, b1, W2, b2, out);
}
